// MambaLayer_81905026335297
// MI455X (gfx1250) — matrix operand order repaired and checked by reading, not compiled, not run on hardware
//
#include <hip/hip_runtime.h>
#include <hip/hip_bf16.h>
#include <math.h>

#define DIM 512
#define D_STATE 16
#define D_CONV 4
#define D_INNER 1024
#define DT_RANK 32
#define B_SZ 2
#define L_SEQ 2048
#define M_TOT (B_SZ * L_SEQ)   /* 4096 rows of activations */

typedef __attribute__((ext_vector_type(16))) __bf16          v16bf;
typedef __attribute__((ext_vector_type(16))) unsigned short  v16u;
typedef __attribute__((ext_vector_type(8)))  float           v8f;

__device__ __forceinline__ unsigned short f2bf(float f) {
  union { float f; unsigned int u; } c; c.f = f;
  unsigned int u = c.u + 0x7FFFu + ((c.u >> 16) & 1u);   // round-to-nearest-even
  return (unsigned short)(u >> 16);
}
__device__ __forceinline__ float silu_f(float x)     { return x / (1.f + __expf(-x)); }
__device__ __forceinline__ float softplus_f(float x) { return (x > 20.f) ? x : log1pf(__expf(x)); }

// ---------------------------------------------------------------------------
// f32 -> bf16 (bit pattern in ushort) bulk converter (for weight matrices)
// ---------------------------------------------------------------------------
__global__ void k_f32_to_bf16(const float* __restrict__ src,
                              unsigned short* __restrict__ dst, int n) {
  int i = blockIdx.x * blockDim.x + threadIdx.x;
  if (i < n) dst[i] = f2bf(src[i]);
}

// ---------------------------------------------------------------------------
// LayerNorm1: x is (B, DIM, L); produce xn_bf16 (B*L, DIM) row-major (bf16)
// one block per (b,l); 128 threads, 4 features each
// ---------------------------------------------------------------------------
__global__ void k_ln1(const float* __restrict__ x,
                      const float* __restrict__ w, const float* __restrict__ bta,
                      unsigned short* __restrict__ xn) {
  const int bl = blockIdx.x;              // 0 .. B*L-1
  const int bb = bl / L_SEQ, l = bl % L_SEQ;
  const int t  = threadIdx.x;             // 0..127
  __shared__ float red[128];

  float v[4]; float s = 0.f;
#pragma unroll
  for (int i = 0; i < 4; ++i) {
    int d = t + i * 128;
    v[i] = x[((size_t)bb * DIM + d) * L_SEQ + l];
    s += v[i];
  }
  red[t] = s; __syncthreads();
  for (int off = 64; off > 0; off >>= 1) { if (t < off) red[t] += red[t + off]; __syncthreads(); }
  const float mean = red[0] * (1.f / DIM);
  __syncthreads();

  float s2 = 0.f;
#pragma unroll
  for (int i = 0; i < 4; ++i) { float dv = v[i] - mean; s2 += dv * dv; }
  red[t] = s2; __syncthreads();
  for (int off = 64; off > 0; off >>= 1) { if (t < off) red[t] += red[t + off]; __syncthreads(); }
  const float rstd = rsqrtf(red[0] * (1.f / DIM) + 1e-5f);

#pragma unroll
  for (int i = 0; i < 4; ++i) {
    int d = t + i * 128;
    xn[(size_t)bl * DIM + d] = f2bf((v[i] - mean) * rstd * w[d] + bta[d]);
  }
}

// ---------------------------------------------------------------------------
// WMMA GEMM:  C[M x N] = A[M x K] * W[N x K]^T  (A,W bf16; C f32)
// A row-major (leading dim lda); W row-major (ld = K); C row-major ldc.
// Each wave computes a 16x64 strip: one A fragment per K-step feeds FOUR
// v_wmma_f32_16x16x32_bf16 ops (A-reuse => 4x arithmetic intensity on the
// activation stream). A/B fragments use contiguous k order (k = 16*half + j), not the ISA table order.
// act==1: C = softplus(C + bias[n])   (fused dt_proj epilogue)
// Requires N % 64 == 0 (true for 2048 / 64 / 1024 / 512).
// ---------------------------------------------------------------------------
__global__ void k_gemm_bf16(const unsigned short* __restrict__ A,
                            const unsigned short* __restrict__ W,
                            float* __restrict__ C,
                            int M, int N, int K, int lda, int ldc,
                            const float* __restrict__ bias, int act) {
  const int lane  = threadIdx.x & 31;
  const int strip = blockIdx.x * (blockDim.x >> 5) + (threadIdx.x >> 5);
  const int ns = N >> 6, ms = M >> 4;          // strips: 16 rows x 64 cols
  if (strip >= ms * ns) return;                // whole-wave uniform exit
  const int tm  = strip / ns;
  const int tn0 = (strip % ns) * 4;            // first of 4 consecutive 16-wide n tiles
  const int half = lane >> 4, l16 = lane & 15;

  // A: lanes 0-15 cover K 0..15; lanes 16-31 cover K 16..31
  const unsigned short* Arow = A + (size_t)(tm * 16 + l16) * lda + half * 16;
  // B: lanes 0-15 cover K 0..15; lanes 16-31 cover K 16..31 (per n column = W row)
  const unsigned short* Wr0 = W + (size_t)((tn0 + 0) * 16 + l16) * K + half * 16;
  const unsigned short* Wr1 = Wr0 + (size_t)16 * K;
  const unsigned short* Wr2 = Wr0 + (size_t)32 * K;
  const unsigned short* Wr3 = Wr0 + (size_t)48 * K;

  v8f acc0 = {}, acc1 = {}, acc2 = {}, acc3 = {};
  for (int k0 = 0; k0 < K; k0 += 32) {
    v16u au, bu0, bu1, bu2, bu3;
#pragma unroll
    for (int j = 0; j < 16; ++j) {
      au[j]     = Arow[k0 + j];
    }
#pragma unroll
    for (int j = 0; j < 16; ++j) {
      bu0[j] = Wr0[k0 + j];
      bu1[j] = Wr1[k0 + j];
      bu2[j] = Wr2[k0 + j];
      bu3[j] = Wr3[k0 + j];
    }
    v16bf a  = __builtin_bit_cast(v16bf, au);
    v16bf b0 = __builtin_bit_cast(v16bf, bu0);
    v16bf b1 = __builtin_bit_cast(v16bf, bu1);
    v16bf b2 = __builtin_bit_cast(v16bf, bu2);
    v16bf b3 = __builtin_bit_cast(v16bf, bu3);
    acc0 = __builtin_amdgcn_wmma_f32_16x16x32_bf16(false, a, false, b0, (short)0, acc0, false, false);
    acc1 = __builtin_amdgcn_wmma_f32_16x16x32_bf16(false, a, false, b1, (short)0, acc1, false, false);
    acc2 = __builtin_amdgcn_wmma_f32_16x16x32_bf16(false, a, false, b2, (short)0, acc2, false, false);
    acc3 = __builtin_amdgcn_wmma_f32_16x16x32_bf16(false, a, false, b3, (short)0, acc3, false, false);
  }

  // C/D layout: VGPR i -> row (i + 8*half), lane&15 -> column within tile
#pragma unroll
  for (int t = 0; t < 4; ++t) {
    v8f acc = (t == 0) ? acc0 : (t == 1) ? acc1 : (t == 2) ? acc2 : acc3;
    const int n = (tn0 + t) * 16 + l16;
    const float bv = (act == 1) ? bias[n] : 0.f;
#pragma unroll
    for (int i = 0; i < 8; ++i) {
      int m = tm * 16 + i + half * 8;
      float v = acc[i];
      if (act == 1) v = softplus_f(v + bv);
      C[(size_t)m * ldc + n] = v;
    }
  }
}

// ---------------------------------------------------------------------------
// causal depthwise conv (K=4) + SiLU on x_in = xz[..., :D_INNER]
// outputs xc (f32) and xc_bf16
// ---------------------------------------------------------------------------
__global__ void k_conv_silu(const float* __restrict__ xz,
                            const float* __restrict__ cw, const float* __restrict__ cb,
                            float* __restrict__ xc, unsigned short* __restrict__ xcb) {
  size_t idx = (size_t)blockIdx.x * blockDim.x + threadIdx.x;
  if (idx >= (size_t)M_TOT * D_INNER) return;
  int d = idx % D_INNER;
  int bl = idx / D_INNER;
  int b = bl / L_SEQ, l = bl % L_SEQ;
  float s = cb[d];
#pragma unroll
  for (int k = 0; k < D_CONV; ++k) {
    int lk = l - (D_CONV - 1) + k;
    if (lk >= 0) s += xz[((size_t)(b * L_SEQ + lk)) * (2 * D_INNER) + d] * cw[d * D_CONV + k];
  }
  s = silu_f(s);
  xc[idx] = s;
  xcb[idx] = f2bf(s);
}

// ---------------------------------------------------------------------------
// pack dt_lo = dbl[..., :DT_RANK] (stride 64) into contiguous bf16 (M x 32)
// ---------------------------------------------------------------------------
__global__ void k_pack_dtlo(const float* __restrict__ dbl, unsigned short* __restrict__ dst) {
  int i = blockIdx.x * blockDim.x + threadIdx.x;
  if (i >= M_TOT * DT_RANK) return;
  int m = i / DT_RANK, r = i % DT_RANK;
  dst[i] = f2bf(dbl[(size_t)m * (DT_RANK + 2 * D_STATE) + r]);
}

// ---------------------------------------------------------------------------
// selective scan: lanes 0-15 = states of channel (2*blk), lanes 16-31 = (2*blk+1)
// h kept in registers; dA, dBu recomputed per step; y via half-wave shfl reduce
// ---------------------------------------------------------------------------
__global__ void k_scan(const float* __restrict__ dt,   // (M, D_INNER) softplus'd
                       const float* __restrict__ dbl,  // (M, 64): [.,32:48]=B, [.,48:64]=C
                       const float* __restrict__ xc,   // (M, D_INNER)
                       const float* __restrict__ A_log,
                       float* __restrict__ y) {
  const int lane = threadIdx.x;
  const int n = lane & 15, half = lane >> 4;
  const int ch = blockIdx.x * 2 + half;        // 0 .. B*D_INNER-1
  const int b = ch / D_INNER, d = ch % D_INNER;

  const float Ac = -__expf(A_log[d * D_STATE + n]);
  float h = 0.f;
  const float* dtp  = dt  + (size_t)b * L_SEQ * D_INNER + d;
  const float* xcp  = xc  + (size_t)b * L_SEQ * D_INNER + d;
  const float* dblp = dbl + (size_t)b * L_SEQ * 64;
  float* yp = y + (size_t)b * L_SEQ * D_INNER + d;

  for (int l = 0; l < L_SEQ; ++l) {
    __builtin_prefetch(dtp + (size_t)(l + 16) * D_INNER, 0, 1);   // global_prefetch_b8
    float dtv = dtp[(size_t)l * D_INNER];
    float xv  = xcp[(size_t)l * D_INNER];
    float Bv  = dblp[l * 64 + DT_RANK + n];
    float Cv  = dblp[l * 64 + DT_RANK + D_STATE + n];
    h = __expf(dtv * Ac) * h + dtv * Bv * xv;
    float p = h * Cv;
    p += __shfl_xor(p, 1);
    p += __shfl_xor(p, 2);
    p += __shfl_xor(p, 4);
    p += __shfl_xor(p, 8);
    if (n == 0) yp[(size_t)l * D_INNER] = p;
  }
}

// ---------------------------------------------------------------------------
// y2 = (y + xc*D) * silu(z)   -> bf16 for out_proj
// ---------------------------------------------------------------------------
__global__ void k_gate(const float* __restrict__ y, const float* __restrict__ xc,
                       const float* __restrict__ Dv, const float* __restrict__ xz,
                       unsigned short* __restrict__ y2b) {
  size_t idx = (size_t)blockIdx.x * blockDim.x + threadIdx.x;
  if (idx >= (size_t)M_TOT * D_INNER) return;
  int d = idx % D_INNER;
  size_t m = idx / D_INNER;
  float yv = y[idx] + xc[idx] * Dv[d];
  float zv = xz[m * (2 * D_INNER) + D_INNER + d];
  y2b[idx] = f2bf(yv * silu_f(zv));
}

// ---------------------------------------------------------------------------
// residual + LayerNorm2 + transpose back to (B, DIM, L)
// ---------------------------------------------------------------------------
__global__ void k_ln2_out(const float* __restrict__ x, const float* __restrict__ xm,
                          const float* __restrict__ w, const float* __restrict__ bta,
                          float* __restrict__ out) {
  const int bl = blockIdx.x;
  const int bb = bl / L_SEQ, l = bl % L_SEQ;
  const int t  = threadIdx.x;
  __shared__ float red[128];

  float v[4]; float s = 0.f;
#pragma unroll
  for (int i = 0; i < 4; ++i) {
    int d = t + i * 128;
    v[i] = x[((size_t)bb * DIM + d) * L_SEQ + l] + xm[(size_t)bl * DIM + d];
    s += v[i];
  }
  red[t] = s; __syncthreads();
  for (int off = 64; off > 0; off >>= 1) { if (t < off) red[t] += red[t + off]; __syncthreads(); }
  const float mean = red[0] * (1.f / DIM);
  __syncthreads();

  float s2 = 0.f;
#pragma unroll
  for (int i = 0; i < 4; ++i) { float dv = v[i] - mean; s2 += dv * dv; }
  red[t] = s2; __syncthreads();
  for (int off = 64; off > 0; off >>= 1) { if (t < off) red[t] += red[t + off]; __syncthreads(); }
  const float rstd = rsqrtf(red[0] * (1.f / DIM) + 1e-5f);

#pragma unroll
  for (int i = 0; i < 4; ++i) {
    int d = t + i * 128;
    out[((size_t)bb * DIM + d) * L_SEQ + l] = (v[i] - mean) * rstd * w[d] + bta[d];
  }
}

// ---------------------------------------------------------------------------
// host orchestration
// ---------------------------------------------------------------------------
static inline size_t alignup(size_t v) { return (v + 255) & ~(size_t)255; }

extern "C" void kernel_launch(void* const* d_in, const int* in_sizes, int n_in,
                              void* d_out, int out_size, void* d_ws, size_t ws_size,
                              hipStream_t stream) {
  const float* x         = (const float*)d_in[0];
  const float* n1w       = (const float*)d_in[1];
  const float* n1b       = (const float*)d_in[2];
  const float* n2w       = (const float*)d_in[3];
  const float* n2b       = (const float*)d_in[4];
  const float* in_proj_w = (const float*)d_in[5];    // (2048, 512)
  const float* conv_w    = (const float*)d_in[6];    // (1024, 4)
  const float* conv_b    = (const float*)d_in[7];
  const float* x_proj_w  = (const float*)d_in[8];    // (64, 1024)
  const float* dt_proj_w = (const float*)d_in[9];    // (1024, 32)
  const float* dt_proj_b = (const float*)d_in[10];
  const float* A_log     = (const float*)d_in[11];   // (1024, 16)
  const float* Dvec      = (const float*)d_in[12];
  const float* out_proj_w= (const float*)d_in[13];   // (512, 1024)
  float* out = (float*)d_out;

  char* p = (char*)d_ws; size_t off = 0;
  auto carve = [&](size_t bytes) { void* q = p + off; off += alignup(bytes); return q; };

  unsigned short* xn_bf   = (unsigned short*)carve((size_t)M_TOT * DIM * 2);
  unsigned short* w_in_bf = (unsigned short*)carve((size_t)2 * D_INNER * DIM * 2);
  unsigned short* w_xp_bf = (unsigned short*)carve((size_t)(DT_RANK + 2 * D_STATE) * D_INNER * 2);
  unsigned short* w_dt_bf = (unsigned short*)carve((size_t)D_INNER * DT_RANK * 2);
  unsigned short* w_out_bf= (unsigned short*)carve((size_t)DIM * D_INNER * 2);
  float*          xz      = (float*)carve((size_t)M_TOT * 2 * D_INNER * 4);
  float*          xc      = (float*)carve((size_t)M_TOT * D_INNER * 4);
  unsigned short* xc_bf   = (unsigned short*)carve((size_t)M_TOT * D_INNER * 2);
  float*          dbl     = (float*)carve((size_t)M_TOT * (DT_RANK + 2 * D_STATE) * 4);
  unsigned short* dtlo_bf = (unsigned short*)carve((size_t)M_TOT * DT_RANK * 2);
  float*          dt      = (float*)carve((size_t)M_TOT * D_INNER * 4);
  float*          yb      = (float*)carve((size_t)M_TOT * D_INNER * 4);
  unsigned short* y2_bf   = (unsigned short*)carve((size_t)M_TOT * D_INNER * 2);
  float*          xm      = (float*)carve((size_t)M_TOT * DIM * 4);
  (void)ws_size; (void)n_in; (void)in_sizes; (void)out_size;

  // weight conversions
  {
    int n;
    n = 2 * D_INNER * DIM;              k_f32_to_bf16<<<(n + 255) / 256, 256, 0, stream>>>(in_proj_w, w_in_bf, n);
    n = (DT_RANK + 2*D_STATE) * D_INNER;k_f32_to_bf16<<<(n + 255) / 256, 256, 0, stream>>>(x_proj_w, w_xp_bf, n);
    n = D_INNER * DT_RANK;              k_f32_to_bf16<<<(n + 255) / 256, 256, 0, stream>>>(dt_proj_w, w_dt_bf, n);
    n = DIM * D_INNER;                  k_f32_to_bf16<<<(n + 255) / 256, 256, 0, stream>>>(out_proj_w, w_out_bf, n);
  }

  // LayerNorm1 -> xn_bf (M x DIM)
  k_ln1<<<M_TOT, 128, 0, stream>>>(x, n1w, n1b, xn_bf);

  // in_proj: (M x 2048) = xn (M x 512) * W^T   [16x64 strips]
  {
    int strips = (M_TOT / 16) * ((2 * D_INNER) / 64);
    k_gemm_bf16<<<(strips + 3) / 4, 128, 0, stream>>>(xn_bf, w_in_bf, xz,
        M_TOT, 2 * D_INNER, DIM, DIM, 2 * D_INNER, nullptr, 0);
  }

  // conv + silu
  {
    size_t n = (size_t)M_TOT * D_INNER;
    k_conv_silu<<<(unsigned)((n + 255) / 256), 256, 0, stream>>>(xz, conv_w, conv_b, xc, xc_bf);
  }

  // x_proj: (M x 64) = xc (M x 1024) * W^T
  {
    int strips = (M_TOT / 16) * ((DT_RANK + 2 * D_STATE) / 64);
    k_gemm_bf16<<<(strips + 3) / 4, 128, 0, stream>>>(xc_bf, w_xp_bf, dbl,
        M_TOT, DT_RANK + 2 * D_STATE, D_INNER, D_INNER, DT_RANK + 2 * D_STATE, nullptr, 0);
  }

  // pack dt_lo, then dt_proj (+bias+softplus fused): (M x 1024), K = 32
  {
    int n = M_TOT * DT_RANK;
    k_pack_dtlo<<<(n + 255) / 256, 256, 0, stream>>>(dbl, dtlo_bf);
    int strips = (M_TOT / 16) * (D_INNER / 64);
    k_gemm_bf16<<<(strips + 3) / 4, 128, 0, stream>>>(dtlo_bf, w_dt_bf, dt,
        M_TOT, D_INNER, DT_RANK, DT_RANK, D_INNER, dt_proj_b, 1);
  }

  // selective scan
  k_scan<<<B_SZ * D_INNER / 2, 32, 0, stream>>>(dt, dbl, xc, A_log, yb);

  // gate: y2 = (y + xc*D) * silu(z)
  {
    size_t n = (size_t)M_TOT * D_INNER;
    k_gate<<<(unsigned)((n + 255) / 256), 256, 0, stream>>>(yb, xc, Dvec, xz, y2_bf);
  }

  // out_proj: (M x 512) = y2 (M x 1024) * W^T
  {
    int strips = (M_TOT / 16) * (DIM / 64);
    k_gemm_bf16<<<(strips + 3) / 4, 128, 0, stream>>>(y2_bf, w_out_bf, xm,
        M_TOT, DIM, D_INNER, D_INNER, DIM, nullptr, 0);
  }

  // residual + LayerNorm2 + transpose -> out (B, DIM, L)
  k_ln2_out<<<M_TOT, 128, 0, stream>>>(x, xm, n2w, n2b, out);
}
